// Attention_19104014533260
// MI455X (gfx1250) — matrix operand order repaired and checked by reading, not compiled, not run on hardware
//
#include <hip/hip_runtime.h>
#include <math.h>

// ---------------- CDNA5 WMMA types ----------------
typedef __attribute__((ext_vector_type(16))) __bf16 v16bf;
typedef __attribute__((ext_vector_type(8)))  float  v8f;

union BF16x16 { v16bf v; __bf16 e[16]; };

#define NB      2
#define DIM     256
#define HEADS   4
#define HEAD_DIM 64
#define KEY_DIM 32
#define HIDDEN  512
#define HW      64
#define NPIX    4096
#define EPS     1e-5f

__device__ __forceinline__ v8f wmma_bf16(v16bf a, v16bf b, v8f c) {
    // D = A(16x32 bf16) x B(32x16 bf16) + C(16x16 f32)
    return __builtin_amdgcn_wmma_f32_16x16x32_bf16(
        /*neg_a=*/false, a, /*neg_b=*/false, b,
        /*c_mod=*/(short)0, c, /*reuse_a=*/false, /*reuse_b=*/false);
}

// A-matrix (16-bit, 16x32) per-lane element -> K index (matches B packing):
//  K = half*16 + e
__device__ __forceinline__ int a16_k(int e, int half) {
    int r = e >> 1, j = e & 1;
    return half * 16 + e;
}

// ---------------------------------------------------------------------------
// Pack fp32 row-major W[M x K] into bf16 WMMA A-fragments:
//   af[ (mt*(K/32)+kb) ][lane][16]   (one 32B contiguous chunk per lane)
// grid = (M/16)*(K/32) blocks of 32.
// ---------------------------------------------------------------------------
__global__ void __launch_bounds__(32)
pack_a_kernel(const float* __restrict__ W, __bf16* __restrict__ af, int M, int K)
{
    const int lane = threadIdx.x & 31;
    const int half = lane >> 4;
    const int col  = lane & 15;
    const int kbc  = K >> 5;
    const int frag = blockIdx.x;          // mt*kbc + kb
    const int kb   = frag % kbc;
    const int mt   = frag / kbc;
    const int m    = mt * 16 + col;
    BF16x16 a;
#pragma unroll
    for (int e = 0; e < 16; ++e) {
        int k = kb * 32 + a16_k(e, half);
        a.e[e] = (__bf16)W[(size_t)m * K + k];
    }
    *(v16bf*)(af + ((size_t)frag * 32 + lane) * 16) = a.v;
}

// ---------------------------------------------------------------------------
// Pack fp32 X[NB x K x N] into bf16 WMMA B-fragments:
//   bf[ ((b*(K/32)+kb)*(N/16)+nt) ][lane][16]
// grid = NB*(K/32)*(N/16) blocks of 32.
// ---------------------------------------------------------------------------
__global__ void __launch_bounds__(32)
pack_b_kernel(const float* __restrict__ X, __bf16* __restrict__ bfrag, int K, int N)
{
    const int lane = threadIdx.x & 31;
    const int half = lane >> 4;
    const int col  = lane & 15;
    const int ntc  = N >> 4, kbc = K >> 5;
    int frag = blockIdx.x;
    const int nt = frag % ntc;
    const int kb = (frag / ntc) % kbc;
    const int b  = frag / (ntc * kbc);
    const int n  = nt * 16 + col;
    BF16x16 bb;
#pragma unroll
    for (int e = 0; e < 16; ++e) {
        int k = kb * 32 + half * 16 + e;       // B layout: K = half*16 + e
        bb.e[e] = (__bf16)X[((size_t)b * K + k) * N + n];
    }
    *(v16bf*)(bfrag + ((size_t)frag * 32 + lane) * 16) = bb.v;
}

// ---------------------------------------------------------------------------
// Fragment GEMM + folded BN:  Out[b][m][n] = BN( sum_k W[m][k] * X[b][k][n] )
// One wave owns a 16(m) x 64(n) tile: A fragment reused across 4 WMMAs/k-step.
// ---------------------------------------------------------------------------
__global__ void __launch_bounds__(32)
gemm_frag_bn_kernel(const __bf16* __restrict__ AF, const __bf16* __restrict__ BF,
                    const float* __restrict__ g, const float* __restrict__ bt,
                    const float* __restrict__ mu, const float* __restrict__ var,
                    float* __restrict__ Out, int M, int K, int N)
{
    const int lane = threadIdx.x & 31;
    const int half = lane >> 4;
    const int col  = lane & 15;
    const int kbc = K >> 5, ntc = N >> 4, n64 = N >> 6, mtc = M >> 4;

    int tile = blockIdx.x;
    const int b   = tile / (mtc * n64);
    int rem       = tile % (mtc * n64);
    const int mt  = rem / n64;
    const int nt0 = (rem % n64) * 4;

    v8f acc[4];
    v8f zero = {};
#pragma unroll
    for (int t = 0; t < 4; ++t) acc[t] = zero;

    const __bf16* ap = AF + ((size_t)mt * kbc * 32 + lane) * 16;
    const __bf16* bp = BF + (((size_t)b * kbc * ntc + nt0) * 32 + lane) * 16;

    for (int kb = 0; kb < kbc; ++kb) {
        v16bf a = *(const v16bf*)(ap + (size_t)kb * 32 * 16);
        __builtin_prefetch(ap + (size_t)(kb + 1) * 32 * 16, 0, 1);
        __builtin_prefetch(bp + (size_t)(kb + 1) * ntc * 32 * 16, 0, 1);
#pragma unroll
        for (int t = 0; t < 4; ++t) {
            v16bf bv = *(const v16bf*)(bp + ((size_t)kb * ntc + t) * 32 * 16);
            acc[t] = wmma_bf16(a, bv, acc[t]);
        }
    }

    // BN epilogue: rows of C tile are (r + 8*half)
#pragma unroll
    for (int r = 0; r < 8; ++r) {
        const int mm = mt * 16 + r + 8 * half;
        const float sc = g[mm] * rsqrtf(var[mm] + EPS);
        const float bi = bt[mm] - mu[mm] * sc;
#pragma unroll
        for (int t = 0; t < 4; ++t)
            Out[((size_t)b * M + mm) * N + (nt0 + t) * 16 + col] = acc[t][r] * sc + bi;
    }
}

// ---------------------------------------------------------------------------
// Pack Q tiles as WMMA A-fragments (m=i, k=d) and K tiles as B-fragments
// (k=d, n=j) so the attention loop loads one contiguous 32B vector per lane.
// grid = NB*HEADS*256 fragments, 64 threads (wave0 -> Q, wave1 -> K).
// ---------------------------------------------------------------------------
__global__ void __launch_bounds__(64)
prep_qk_kernel(const float* __restrict__ qkv,   // [NB x 512 x 4096]
               __bf16* __restrict__ qf, __bf16* __restrict__ kf)
{
    const int frag = blockIdx.x;               // (b*4 + h)*256 + t
    const int t = frag & 255;
    const int h = (frag >> 8) & 3;
    const int b = frag >> 10;
    const int lane = threadIdx.x & 31;
    const int half = lane >> 4;
    const int col  = lane & 15;
    const float* base = qkv + ((size_t)b * HIDDEN + h * 128) * NPIX;

    if (threadIdx.x < 32) {                    // Q fragment: A(m=i, k=d) = q[d][i]
        BF16x16 a;
        const int i = t * 16 + col;
#pragma unroll
        for (int e = 0; e < 16; ++e) {
            int d = a16_k(e, half);
            a.e[e] = (__bf16)base[(size_t)d * NPIX + i];
        }
        *(v16bf*)(qf + ((size_t)frag * 32 + lane) * 16) = a.v;
    } else {                                   // K fragment: B(k=d, n=j) = k[d][j]
        BF16x16 bb;
        const int j = t * 16 + col;
#pragma unroll
        for (int e = 0; e < 16; ++e) {
            int d = half * 16 + e;
            bb.e[e] = (__bf16)base[(size_t)(32 + d) * NPIX + j];
        }
        *(v16bf*)(kf + ((size_t)frag * 32 + lane) * 16) = bb.v;
    }
}

// v (fp32 planes inside qkv) -> bf16 planar [ (b*4+h)*64 + d ][ j ]
__global__ void cvt_v_kernel(const float* __restrict__ qkv, __bf16* __restrict__ vbf)
{
    size_t idx = (size_t)blockIdx.x * 256 + threadIdx.x;   // NB*4*64*4096 total
    int j = (int)(idx & 4095);
    int p = (int)(idx >> 12);
    int d = p & 63, h = (p >> 6) & 3, b = p >> 8;
    vbf[idx] = (__bf16)qkv[((size_t)b * HIDDEN + h * 128 + 64 + d) * NPIX + j];
}

// ---------------------------------------------------------------------------
// Flash attention: one wave per (b, h, 16-row query tile). j-blocks of 32:
//   2x S-WMMA, shuffle row-max/sum (rows align with accumulator VGPRs),
//   probs -> LDS(bf16) to convert C-layout -> A-layout, 4x PV-WMMA.
// ---------------------------------------------------------------------------
__global__ void __launch_bounds__(32)
attn_kernel(const __bf16* __restrict__ qf, const __bf16* __restrict__ kf,
            const __bf16* __restrict__ vbf, float* __restrict__ y)
{
    __shared__ __bf16 p_lds[16][32];           // 1KB probability tile
    const int lane = threadIdx.x & 31;
    const int half = lane >> 4;
    const int col  = lane & 15;
    const int frag = blockIdx.x;               // (b*4 + h)*256 + it
    const int it = frag & 255;
    const int h  = (frag >> 8) & 3;
    const int b  = frag >> 10;

    const v16bf qa = *(const v16bf*)(qf + ((size_t)frag * 32 + lane) * 16);
    const __bf16* kbase = kf + ((size_t)(b * 4 + h) * 256) * 32 * 16;
    const __bf16* vbase = vbf + (size_t)(b * 4 + h) * 64 * NPIX;

    v8f yacc[4];
    v8f zero = {};
#pragma unroll
    for (int dt = 0; dt < 4; ++dt) yacc[dt] = zero;
    float mrun[8], lrun[8];
#pragma unroll
    for (int r = 0; r < 8; ++r) { mrun[r] = -INFINITY; lrun[r] = 0.f; }
    const float scale = 0.17677669529663687f;  // KEY_DIM^-0.5

    for (int j0 = 0; j0 < NPIX; j0 += 32) {
        const int jt = j0 >> 4;
        v16bf kb0 = *(const v16bf*)(kbase + ((size_t)jt * 32 + lane) * 16);
        v16bf kb1 = *(const v16bf*)(kbase + ((size_t)(jt + 1) * 32 + lane) * 16);
        // prefetch next j-block's K fragments and V rows (global_prefetch_b8)
        __builtin_prefetch(kbase + ((size_t)(jt + 2) * 32 + lane) * 16, 0, 1);
        __builtin_prefetch(vbase + (size_t)col * NPIX + j0 + 32 + half * 16, 0, 1);
        v8f s0 = wmma_bf16(qa, kb0, zero);
        v8f s1 = wmma_bf16(qa, kb1, zero);

        float alpha[8];
#pragma unroll
        for (int r = 0; r < 8; ++r) {
            float sv0 = s0[r] * scale, sv1 = s1[r] * scale;
            float mx = fmaxf(sv0, sv1);
            mx = fmaxf(mx, __shfl_xor(mx, 1));
            mx = fmaxf(mx, __shfl_xor(mx, 2));
            mx = fmaxf(mx, __shfl_xor(mx, 4));
            mx = fmaxf(mx, __shfl_xor(mx, 8));   // row max over 16 lanes of the half
            float mnew = fmaxf(mrun[r], mx);
            float al = __expf(mrun[r] - mnew);
            float p0 = __expf(sv0 - mnew);
            float p1 = __expf(sv1 - mnew);
            float rs = p0 + p1;
            rs += __shfl_xor(rs, 1);
            rs += __shfl_xor(rs, 2);
            rs += __shfl_xor(rs, 4);
            rs += __shfl_xor(rs, 8);
            lrun[r] = lrun[r] * al + rs;
            mrun[r] = mnew;
            alpha[r] = al;
            p_lds[r + 8 * half][col]      = (__bf16)p0;
            p_lds[r + 8 * half][16 + col] = (__bf16)p1;
        }
#pragma unroll
        for (int dt = 0; dt < 4; ++dt)
#pragma unroll
            for (int r = 0; r < 8; ++r) yacc[dt][r] *= alpha[r];

        __syncthreads();
        BF16x16 pa;                             // C-layout -> A-layout (m=i,k=j)
#pragma unroll
        for (int e = 0; e < 16; ++e) pa.e[e] = p_lds[col][a16_k(e, half)];
        __syncthreads();

#pragma unroll
        for (int dt = 0; dt < 4; ++dt) {
            // B(k=j, n=d) = v[d][j] : contiguous 16 bf16 in j per lane
            const __bf16* vp = vbase + (size_t)(dt * 16 + col) * NPIX + j0 + half * 16;
            v16bf vb = *(const v16bf*)vp;
            yacc[dt] = wmma_bf16(pa.v, vb, yacc[dt]);
        }
    }

    // y_t[m=i][n=d] -> y[(b*256 + h*64 + d)][it*16 + i]; 8 consecutive floats/lane
#pragma unroll
    for (int dt = 0; dt < 4; ++dt)
#pragma unroll
        for (int r = 0; r < 8; ++r) {
            float val = yacc[dt][r] / lrun[r];
            y[((size_t)b * DIM + h * 64 + dt * 16 + col) * NPIX + it * 16 + 8 * half + r] = val;
        }
}

// ---------------------------------------------------------------------------
// Depthwise 3x3 on v + BN, accumulated into y:  y += BN(dwconv(v))
// ---------------------------------------------------------------------------
__global__ void pe_add_kernel(const float* __restrict__ qkv, const float* __restrict__ pw,
                              const float* __restrict__ g, const float* __restrict__ bt,
                              const float* __restrict__ mu, const float* __restrict__ var,
                              float* __restrict__ y)
{
    size_t idx = (size_t)blockIdx.x * 256 + threadIdx.x;   // NB*256*4096
    int px = (int)(idx & 4095);
    int c  = (int)(idx >> 12) & 255;
    int b  = (int)(idx >> 20);
    int xx = px & 63, yy = px >> 6;
    int h = c >> 6, d = c & 63;
    const float* vpl = qkv + ((size_t)b * HIDDEN + h * 128 + 64 + d) * NPIX;
    const float* w = pw + c * 9;
    float acc = 0.f;
#pragma unroll
    for (int dy = -1; dy <= 1; ++dy)
#pragma unroll
        for (int dx = -1; dx <= 1; ++dx) {
            int yv = yy + dy, xv = xx + dx;
            if (yv >= 0 && yv < HW && xv >= 0 && xv < HW)
                acc += w[(dy + 1) * 3 + (dx + 1)] * vpl[yv * HW + xv];
        }
    float sc = g[c] * rsqrtf(var[c] + EPS);
    y[idx] += acc * sc + (bt[c] - mu[c] * sc);
}

// ---------------------------------------------------------------------------
extern "C" void kernel_launch(void* const* d_in, const int* in_sizes, int n_in,
                              void* d_out, int out_size, void* d_ws, size_t ws_size,
                              hipStream_t stream)
{
    const float* x      = (const float*)d_in[0];
    const float* qkv_w  = (const float*)d_in[1];
    const float* qkv_g  = (const float*)d_in[2];
    const float* qkv_b  = (const float*)d_in[3];
    const float* qkv_m  = (const float*)d_in[4];
    const float* qkv_v  = (const float*)d_in[5];
    const float* proj_w = (const float*)d_in[6];
    const float* proj_g = (const float*)d_in[7];
    const float* proj_b = (const float*)d_in[8];
    const float* proj_m = (const float*)d_in[9];
    const float* proj_v = (const float*)d_in[10];
    const float* pe_w   = (const float*)d_in[11];
    const float* pe_g   = (const float*)d_in[12];
    const float* pe_b   = (const float*)d_in[13];
    const float* pe_m   = (const float*)d_in[14];
    const float* pe_v   = (const float*)d_in[15];
    float* out = (float*)d_out;

    char* ws = (char*)d_ws;
    const size_t MB = 1024 * 1024;
    float*  qkv_f32 = (float*)(ws);                    // 16 MB : qkv activations
    __bf16* qf      = (__bf16*)(ws + 16 * MB);         //  2 MB : Q A-fragments
    __bf16* kf      = (__bf16*)(ws + 18 * MB);         //  2 MB : K B-fragments
    __bf16* vbf     = (__bf16*)(ws + 20 * MB);         //  4 MB : V bf16 planar
    float*  ybuf    = (float*)(ws + 24 * MB);          //  8 MB : attention out + pe
    __bf16* xbf     = (__bf16*)(ws + 32 * MB);         //  4 MB : x B-fragments
    __bf16* ybf     = (__bf16*)(ws + 36 * MB);         //  4 MB : y B-fragments
    __bf16* awq     = (__bf16*)(ws + 40 * MB);         // 256KB : qkv_w A-fragments
    __bf16* awp     = (__bf16*)(ws + 40 * MB + 512 * 1024); // 128KB : proj_w A-frags

    // 0) pack weights + input activations into WMMA fragments
    pack_a_kernel<<<(HIDDEN / 16) * (DIM / 32), 32, 0, stream>>>(qkv_w, awq, HIDDEN, DIM);
    pack_a_kernel<<<(DIM / 16) * (DIM / 32), 32, 0, stream>>>(proj_w, awp, DIM, DIM);
    pack_b_kernel<<<NB * (DIM / 32) * (NPIX / 16), 32, 0, stream>>>(x, xbf, DIM, NPIX);

    // 1) qkv = BN(W_qkv @ x):   M=512, K=256, N=4096, 2 batches (16x64 tile/wave)
    gemm_frag_bn_kernel<<<NB * (HIDDEN / 16) * (NPIX / 64), 32, 0, stream>>>(
        awq, xbf, qkv_g, qkv_b, qkv_m, qkv_v, qkv_f32, HIDDEN, DIM, NPIX);

    // 2) pack Q/K WMMA fragments, 3) v -> bf16
    prep_qk_kernel<<<NB * HEADS * (NPIX / 16), 64, 0, stream>>>(qkv_f32, qf, kf);
    cvt_v_kernel<<<(NB * HEADS * HEAD_DIM * NPIX) / 256, 256, 0, stream>>>(qkv_f32, vbf);

    // 4) flash attention -> ybuf
    attn_kernel<<<NB * HEADS * (NPIX / 16), 32, 0, stream>>>(qf, kf, vbf, ybuf);

    // 5) ybuf += BN(dwconv3x3(v))
    pe_add_kernel<<<(NB * DIM * NPIX) / 256, 256, 0, stream>>>(
        qkv_f32, pe_w, pe_g, pe_b, pe_m, pe_v, ybuf);

    // 6) pack y, then out = BN(W_proj @ ybuf): M=256, K=256, N=4096
    pack_b_kernel<<<NB * (DIM / 32) * (NPIX / 16), 32, 0, stream>>>(ybuf, ybf, DIM, NPIX);
    gemm_frag_bn_kernel<<<NB * (DIM / 16) * (NPIX / 64), 32, 0, stream>>>(
        awp, ybf, proj_g, proj_b, proj_m, proj_v, out, DIM, DIM, NPIX);
}
